// singlePerspectiveNet_22514218566052
// MI455X (gfx1250) — matrix operand order repaired and checked by reading, not compiled, not run on hardware
//
#include <hip/hip_runtime.h>
#include <stdint.h>
#include <stddef.h>

// ---------------------------------------------------------------------------
// Types for CDNA5 WMMA bf16 fragments
// ---------------------------------------------------------------------------
typedef __attribute__((ext_vector_type(16))) __bf16 v16bf;
typedef __attribute__((ext_vector_type(8)))  float  v8f;

struct alignas(16) U4 { unsigned x, y, z, w; };

union Frag {
    v16bf v;
    U4    q[2];
};

__device__ __forceinline__ unsigned short f2bf(float f) {
    union { float f; unsigned u; } x;
    x.f = f;
    unsigned r = (x.u + 0x7FFFu + ((x.u >> 16) & 1u)) >> 16;  // round-nearest-even
    return (unsigned short)r;
}

__device__ __forceinline__ v8f wmma_bf16(const Frag& a, const Frag& b, v8f c) {
    // (neg_a, A, neg_b, B, c_mod, C, reuse_a, reuse_b)
    return __builtin_amdgcn_wmma_f32_16x16x32_bf16(false, a.v, false, b.v,
                                                   (short)0, c, false, false);
}

// A-fragment (16x32 bf16): lane<16 -> row=lane, K runs {k0..k0+7, k0+16..k0+23};
// lane>=16 -> row=lane-16, runs shifted by 8.  => two aligned 16B loads per lane.
__device__ __forceinline__ Frag loadA(const unsigned short* base, int stride,
                                      int row, int k0, int half) {
    const unsigned short* p = base + (size_t)row * stride + k0 + half * 8;
    Frag f;
    f.q[0] = *(const U4*)p;
    f.q[1] = *(const U4*)(p + 16);
    return f;
}

// B-fragment (32x16 bf16): lane holds column n = n0+lane%16, K runs
// {k0+(lane/16)*8 .. +7, +16 .. +23} => two aligned 16B loads per lane.
__device__ __forceinline__ Frag loadB(const unsigned short* base, int stride,
                                      int row, int k0, int half) {
    const unsigned short* p = base + (size_t)row * stride + k0 + half * 8;
    Frag f;
    f.q[0] = *(const U4*)p;
    f.q[1] = *(const U4*)(p + 16);
    return f;
}

// ---------------------------------------------------------------------------
// Prep kernels: f32 -> bf16 conversion and gathered conversion
// ---------------------------------------------------------------------------
__global__ void cvt_kernel(const float* __restrict__ src,
                           unsigned short* __restrict__ dst, int n) {
    int i = blockIdx.x * blockDim.x + threadIdx.x;
    if (i < n) dst[i] = f2bf(src[i]);
}

__global__ void gather_kernel(const float* __restrict__ x,
                              const int* __restrict__ idx,
                              unsigned short* __restrict__ dst, int d) {
    int row = blockIdx.x;
    const float* xr = x + (size_t)row * 768;
    unsigned short* dr = dst + (size_t)row * d;
    for (int j = threadIdx.x; j < d; j += blockDim.x)
        dr[j] = f2bf(xr[idx[j]]);
}

// ---------------------------------------------------------------------------
// Fused subnet + residual kernel
// ---------------------------------------------------------------------------
struct SubnetParams {
    const unsigned short* xg;   // gathered bf16 [4096, d]
    const unsigned short* Wft;  // bf16 [2048, d]   (row n = FT output col n)
    const unsigned short* W1;   // bf16 [16, 1024]
    const unsigned short* Wo;   // bf16 [256, 32]
    const float* bft;
    const float* b1;
    const float* bo;
    int d;
    int pad;
};

struct FusedParams {
    SubnetParams sub[4];
    const unsigned short* xb;   // bf16 [4096, 768]
    const unsigned short* Wc;   // bf16 [256, 768]
    float* out;                 // f32 [4096, 256]
};

#define SCALE (127.0f / 128.0f)
#define NWAVE 8

// grid.x = 256 (16-row tiles), block = 256 threads (8 wave32s, ~2 waves/SIMD).
// Each wave handles 128 of the 1024 inter columns for f, then 32 output cols.
__global__ __launch_bounds__(256, 1) void fused_kernel(FusedParams P) {
    const int lane = threadIdx.x & 31;
    const int wave = threadIdx.x >> 5;
    const int l16  = lane & 15;
    const int half = lane >> 4;
    const int row0 = blockIdx.x * 16;

    __shared__ unsigned short sInter[NWAVE][16 * 32];  // per-wave inter tile (bf16)
    __shared__ unsigned short sCat[NWAVE][16 * 32];    // per-wave cat tile  (bf16)
    __shared__ float sF[NWAVE][32][8];                 // f partials, C/D layout

    v8f outAcc[2];
#pragma unroll
    for (int i = 0; i < 2; ++i)
        outAcc[i] = (v8f){0.f, 0.f, 0.f, 0.f, 0.f, 0.f, 0.f, 0.f};

    for (int s = 0; s < 4; ++s) {
        const SubnetParams sp = P.sub[s];
        const int d = sp.d;

        v8f accF = (v8f){0.f, 0.f, 0.f, 0.f, 0.f, 0.f, 0.f, 0.f};
        unsigned short* Li = sInter[wave];

        // ---- feature transform + pairwise product + f accumulation ----
        const int nBeg = wave * 128, nEnd = nBeg + 128;
        for (int n0 = nBeg; n0 < nEnd; n0 += 32) {
            v8f t00 = (v8f){0.f,0.f,0.f,0.f,0.f,0.f,0.f,0.f};
            v8f t01 = t00, t10 = t00, t11 = t00;
            for (int k0 = 0; k0 < d; k0 += 32) {
                Frag a  = loadA(sp.xg, d, row0 + l16, k0, half);
                Frag b0 = loadB(sp.Wft, d, n0 + l16,        k0, half);
                Frag b1v = loadB(sp.Wft, d, n0 + 16 + l16,  k0, half);
                Frag b2 = loadB(sp.Wft, d, 1024 + n0 + l16, k0, half);
                Frag b3 = loadB(sp.Wft, d, 1040 + n0 + l16, k0, half);
                t00 = wmma_bf16(a, b0,  t00);
                t01 = wmma_bf16(a, b1v, t01);
                t10 = wmma_bf16(a, b2,  t10);
                t11 = wmma_bf16(a, b3,  t11);
            }
            // biases for this lane's columns
            float bf00 = sp.bft[n0 + l16];
            float bf01 = sp.bft[n0 + 16 + l16];
            float bf10 = sp.bft[1024 + n0 + l16];
            float bf11 = sp.bft[1040 + n0 + l16];
            // inter = (t0+b)*(t1+b)*SCALE, relayout C/D -> A via LDS (bf16)
#pragma unroll
            for (int r = 0; r < 8; ++r) {
                int m = r + 8 * half;
                float i0 = (t00[r] + bf00) * (t10[r] + bf10) * SCALE;
                float i1 = (t01[r] + bf01) * (t11[r] + bf11) * SCALE;
                Li[m * 32 + l16]      = f2bf(i0);
                Li[m * 32 + 16 + l16] = f2bf(i1);
            }
            asm volatile("s_wait_dscnt 0" ::: "memory");  // intra-wave cross-lane LDS hazard
            // A from LDS inter tile
            Frag ia;
            {
                const unsigned short* p = Li + l16 * 32 + half * 8;
                ia.q[0] = *(const U4*)p;
                ia.q[1] = *(const U4*)(p + 16);
            }
            Frag w1f = loadB(sp.W1, 1024, l16, n0, half);
            accF = wmma_bf16(ia, w1f, accF);
        }

        // ---- cross-wave reduction of f ----
#pragma unroll
        for (int r = 0; r < 8; ++r) sF[wave][lane][r] = accF[r];
        __syncthreads();

        float fv[8];
#pragma unroll
        for (int r = 0; r < 8; ++r) {
            float acc = sp.b1[l16];
#pragma unroll
            for (int w = 0; w < NWAVE; ++w) acc += sF[w][lane][r];
            fv[r] = acc;
        }

        // ---- cat = clip([f, f*f*SCALE], 0, 1) as A fragment via LDS ----
        unsigned short* Lc = sCat[wave];
#pragma unroll
        for (int r = 0; r < 8; ++r) {
            int m = r + 8 * half;
            float c0 = fminf(fmaxf(fv[r], 0.f), 1.f);
            float c1 = fminf(fmaxf(fv[r] * fv[r] * SCALE, 0.f), 1.f);
            Lc[m * 32 + l16]      = f2bf(c0);
            Lc[m * 32 + 16 + l16] = f2bf(c1);
        }
        asm volatile("s_wait_dscnt 0" ::: "memory");
        Frag ca;
        {
            const unsigned short* p = Lc + l16 * 32 + half * 8;
            ca.q[0] = *(const U4*)p;
            ca.q[1] = *(const U4*)(p + 16);
        }
        // ---- out += cat @ Wo.T  (wave owns cols [wave*32, wave*32+32)) ----
#pragma unroll
        for (int nt = 0; nt < 2; ++nt) {
            int col = wave * 32 + nt * 16 + l16;
            Frag wo = loadB(sp.Wo, 32, col, 0, half);
            outAcc[nt] = wmma_bf16(ca, wo, outAcc[nt]);
        }
        __syncthreads();  // protect sF before next subnet reuses it
    }

    // ---- residual: out += x @ Wc.T ----
    for (int k0 = 0; k0 < 768; k0 += 32) {
        Frag a = loadA(P.xb, 768, row0 + l16, k0, half);
#pragma unroll
        for (int nt = 0; nt < 2; ++nt) {
            int col = wave * 32 + nt * 16 + l16;
            Frag b = loadB(P.Wc, 768, col, k0, half);
            outAcc[nt] = wmma_bf16(a, b, outAcc[nt]);
        }
    }

    // ---- add output biases and store ----
#pragma unroll
    for (int nt = 0; nt < 2; ++nt) {
        int col = wave * 32 + nt * 16 + l16;
        float bsum = P.sub[0].bo[col] + P.sub[1].bo[col] +
                     P.sub[2].bo[col] + P.sub[3].bo[col];
#pragma unroll
        for (int r = 0; r < 8; ++r) {
            int row = row0 + r + 8 * half;
            P.out[(size_t)row * 256 + col] = outAcc[nt][r] + bsum;
        }
    }
}

// ---------------------------------------------------------------------------
// Host launcher
// ---------------------------------------------------------------------------
extern "C" void kernel_launch(void* const* d_in, const int* in_sizes, int n_in,
                              void* d_out, int out_size, void* d_ws, size_t ws_size,
                              hipStream_t stream) {
    (void)in_sizes; (void)n_in; (void)out_size; (void)ws_size;

    const float* x = (const float*)d_in[0];
    const int* idx[4] = {(const int*)d_in[1], (const int*)d_in[2],
                         (const int*)d_in[3], (const int*)d_in[4]};
    const int dsub[4] = {128, 256, 384, 384};

    const float *Wft[4], *bft[4], *W1[4], *b1[4], *Wo[4], *bo[4];
    for (int s = 0; s < 4; ++s) {
        int b = 5 + s * 6;
        Wft[s] = (const float*)d_in[b + 0];
        bft[s] = (const float*)d_in[b + 1];
        W1[s]  = (const float*)d_in[b + 2];
        b1[s]  = (const float*)d_in[b + 3];
        Wo[s]  = (const float*)d_in[b + 4];
        bo[s]  = (const float*)d_in[b + 5];
    }
    const float* Wc = (const float*)d_in[29];

    // workspace layout (256B aligned regions)
    char* ws = (char*)d_ws;
    size_t off = 0;
    auto alloc = [&](size_t bytes) -> char* {
        off = (off + 255) & ~(size_t)255;
        char* p = ws + off;
        off += bytes;
        return p;
    };
    unsigned short* xb = (unsigned short*)alloc((size_t)4096 * 768 * 2);
    unsigned short *xg[4], *Wftb[4], *W1b[4], *Wob[4];
    for (int s = 0; s < 4; ++s) {
        xg[s]   = (unsigned short*)alloc((size_t)4096 * dsub[s] * 2);
        Wftb[s] = (unsigned short*)alloc((size_t)2048 * dsub[s] * 2);
        W1b[s]  = (unsigned short*)alloc((size_t)16 * 1024 * 2);
        Wob[s]  = (unsigned short*)alloc((size_t)256 * 32 * 2);
    }
    unsigned short* Wcb = (unsigned short*)alloc((size_t)256 * 768 * 2);

    auto cvt = [&](const float* src, unsigned short* dst, int n) {
        cvt_kernel<<<(n + 255) / 256, 256, 0, stream>>>(src, dst, n);
    };

    cvt(x, xb, 4096 * 768);
    for (int s = 0; s < 4; ++s) {
        cvt(Wft[s], Wftb[s], 2048 * dsub[s]);
        cvt(W1[s], W1b[s], 16 * 1024);
        cvt(Wo[s], Wob[s], 256 * 32);
        gather_kernel<<<4096, 128, 0, stream>>>(x, idx[s], xg[s], dsub[s]);
    }
    cvt(Wc, Wcb, 256 * 768);

    FusedParams P;
    for (int s = 0; s < 4; ++s) {
        P.sub[s].xg  = xg[s];
        P.sub[s].Wft = Wftb[s];
        P.sub[s].W1  = W1b[s];
        P.sub[s].Wo  = Wob[s];
        P.sub[s].bft = bft[s];
        P.sub[s].b1  = b1[s];
        P.sub[s].bo  = bo[s];
        P.sub[s].d   = dsub[s];
        P.sub[s].pad = 0;
    }
    P.xb = xb;
    P.Wc = Wcb;
    P.out = (float*)d_out;

    fused_kernel<<<256, 256, 0, stream>>>(P);
}
